// GaddyRawEMGEncoder_42537356100016
// MI455X (gfx1250) — matrix operand order repaired and checked by reading, not compiled, not run on hardware
//
#include <hip/hip_runtime.h>

typedef __attribute__((ext_vector_type(16))) __bf16 v16bf;
typedef __attribute__((ext_vector_type(8)))  __bf16 v8bf;
typedef __attribute__((ext_vector_type(8)))  float  v8f;
typedef unsigned short u16;

__device__ __forceinline__ u16 f2bf(float f) {
  unsigned int u = __float_as_uint(f);
  u += 0x7FFFu + ((u >> 16) & 1u);           // round-to-nearest-even
  return (u16)(u >> 16);
}

// ---------------------------------------------------------------------------
// Generic bf16 WMMA GEMM:  C[z](M x N) = A[z](M x K) * B[z](K x N)  (+epilogue)
//   A element: Ab + m*ldaM + k          (k-stride always 1 in this model)
//   B element: Bb + k*ldbK + n*ldbN
//   C element: Cb + m*ldcM + n
//   z = blockIdx.z; zb = z/H, zh = z%H;  base += zb*s?b + zh*s?h
// Epilogue: v = acc*rowScale[m] + rowShift[m] (+colBias[n]) (+Res[same idx]) (relu)
// Tile: 64x64, 8 waves (wave32), BK=64 (two 32-halves), 4 WMMAs per barrier.
// Interior tiles take fully vectorized staging (b128 global + b128 LDS for A).
// ---------------------------------------------------------------------------
__global__ __launch_bounds__(256) void gemm_bf16_wmma(
    const u16* __restrict__ A, const u16* __restrict__ B, float* __restrict__ C,
    const float* __restrict__ Res,
    const float* __restrict__ rowScale, const float* __restrict__ rowShift,
    const float* __restrict__ colBias,
    int M, int N, int K,
    long ldaM, long ldbK, long ldbN, long ldcM,
    int H, long sAb, long sAh, long sBb, long sBh, long sCb, long sCh,
    int relu)
{
  __shared__ u16 As[64][72];   // [m][k], pad -> 144B row stride (16B aligned)
  __shared__ u16 Bs[64][72];   // [n][k], K-transposed for contiguous frag loads

  const int z  = blockIdx.z;
  const int zb = z / H, zh = z - zb * H;
  const u16*   Ab = A + (long)zb * sAb + (long)zh * sAh;
  const u16*   Bb = B + (long)zb * sBb + (long)zh * sBh;
  float*       Cb = C + (long)zb * sCb + (long)zh * sCh;
  const float* Rb = Res ? Res + (long)zb * sCb + (long)zh * sCh : (const float*)0;

  const int m0 = blockIdx.x * 64;
  const int n0 = blockIdx.y * 64;
  const int tid  = threadIdx.x;
  const int lane = tid & 31;
  const int w    = tid >> 5;
  const int wm = (w & 3) * 16;   // wave's M offset in tile
  const int wn = (w >> 2) * 32;  // wave's N offset in tile (two 16-wide subtiles)
  const int lr = lane & 15;
  const int lh = lane >> 4;

  const bool intM = (m0 + 64 <= M);
  const bool intN = (n0 + 64 <= N);
  const bool bVec = (ldbN == 1);

  v8f acc0 = {0.f,0.f,0.f,0.f,0.f,0.f,0.f,0.f};
  v8f acc1 = {0.f,0.f,0.f,0.f,0.f,0.f,0.f,0.f};

  const int arow = tid >> 2;          // 0..63
  const int akc  = (tid & 3) * 8;     // 0,8,16,24 within a 32-half
  const int bk   = tid >> 3;          // 0..31 k-row within a 32-half
  const int bnc  = (tid & 7) * 8;     // 0..56

  for (int k0 = 0; k0 < K; k0 += 64) {
    if (k0 + 64 < K) {                 // gfx1250 global_prefetch_b8 path
      __builtin_prefetch(&Ab[(long)(m0 + arow) * ldaM + (k0 + 64 + akc)], 0, 1);
      __builtin_prefetch(&Bb[(long)(k0 + 64 + bk) * ldbK + (long)(n0 + bnc) * ldbN], 0, 1);
    }
    #pragma unroll
    for (int hk = 0; hk < 2; ++hk) {
      const int kb = k0 + hk * 32;
      const bool fullK = (kb + 32 <= K);
      // ----- stage A 64x32 half -----
      if (intM && fullK) {
        const v8bf av = *reinterpret_cast<const v8bf*>(
            &Ab[(long)(m0 + arow) * ldaM + (kb + akc)]);
        *reinterpret_cast<v8bf*>(&As[arow][hk * 32 + akc]) = av;
      } else {
        const int gm  = m0 + arow;
        const int gmc = gm < M ? gm : M - 1;
        #pragma unroll
        for (int e = 0; e < 8; ++e) {
          const int gk  = kb + akc + e;
          const int gkc = gk < K ? gk : K - 1;
          u16 v = Ab[(long)gmc * ldaM + gkc];      // clamped, branch-free
          if (gm >= M || gk >= K) v = 0;
          As[arow][hk * 32 + akc + e] = v;
        }
      }
      // ----- stage B 32x64 half (transposed into Bs[n][k]) -----
      if (bVec && intN && fullK) {
        const v8bf bv = *reinterpret_cast<const v8bf*>(
            &Bb[(long)(kb + bk) * ldbK + (n0 + bnc)]);
        #pragma unroll
        for (int e = 0; e < 8; ++e) Bs[bnc + e][hk * 32 + bk] = bv[e];
      } else {
        const int gk  = kb + bk;
        const int gkc = gk < K ? gk : K - 1;
        #pragma unroll
        for (int e = 0; e < 8; ++e) {
          const int gn  = n0 + bnc + e;
          const int gnc = gn < N ? gn : N - 1;
          u16 v = Bb[(long)gkc * ldbK + (long)gnc * ldbN];
          if (gk >= K || gn >= N) v = 0;
          Bs[bnc + e][hk * 32 + bk] = v;
        }
      }
    }
    __syncthreads();

    // A fragments: lane half lh -> k = lh*16 + e (contiguous, matches B)
    const int am = wm + lr;
    v16bf a0, a1;
    {
      const v8bf lo0 = *reinterpret_cast<const v8bf*>(&As[am][lh * 16]);
      const v8bf hi0 = *reinterpret_cast<const v8bf*>(&As[am][8 + lh * 16]);
      const v8bf lo1 = *reinterpret_cast<const v8bf*>(&As[am][32 + lh * 16]);
      const v8bf hi1 = *reinterpret_cast<const v8bf*>(&As[am][40 + lh * 16]);
      #pragma unroll
      for (int e = 0; e < 8; ++e) {
        a0[e] = lo0[e]; a0[8 + e] = hi0[e];
        a1[e] = lo1[e]; a1[8 + e] = hi1[e];
      }
    }
    // B fragments: lane holds column n, k = lh*16 + e (contiguous in Bs[n][..])
    const v16bf b00 = *reinterpret_cast<const v16bf*>(&Bs[wn + lr][lh * 16]);
    const v16bf b01 = *reinterpret_cast<const v16bf*>(&Bs[wn + lr][32 + lh * 16]);
    const v16bf b10 = *reinterpret_cast<const v16bf*>(&Bs[wn + 16 + lr][lh * 16]);
    const v16bf b11 = *reinterpret_cast<const v16bf*>(&Bs[wn + 16 + lr][32 + lh * 16]);

    acc0 = __builtin_amdgcn_wmma_f32_16x16x32_bf16(false, a0, false, b00, (short)0, acc0, false, false);
    acc1 = __builtin_amdgcn_wmma_f32_16x16x32_bf16(false, a0, false, b10, (short)0, acc1, false, false);
    acc0 = __builtin_amdgcn_wmma_f32_16x16x32_bf16(false, a1, false, b01, (short)0, acc0, false, false);
    acc1 = __builtin_amdgcn_wmma_f32_16x16x32_bf16(false, a1, false, b11, (short)0, acc1, false, false);
    __syncthreads();
  }

  // ---------------- epilogue ----------------
  if (intM && intN) {
    #pragma unroll
    for (int r = 0; r < 8; ++r) {
      const int gm = m0 + wm + r + lh * 8;   // C layout: VGPR r -> m = r + 8*(lane/16)
      const float sc = rowScale ? rowScale[gm] : 1.f;
      const float sh = rowShift ? rowShift[gm] : 0.f;
      const int gn0 = n0 + wn + lr;
      const int gn1 = gn0 + 16;
      float v0 = acc0[r] * sc + sh;
      float v1 = acc1[r] * sc + sh;
      if (colBias) { v0 += colBias[gn0]; v1 += colBias[gn1]; }
      const long ci0 = (long)gm * ldcM + gn0;
      const long ci1 = (long)gm * ldcM + gn1;
      if (Rb) { v0 += Rb[ci0]; v1 += Rb[ci1]; }
      if (relu) { v0 = fmaxf(v0, 0.f); v1 = fmaxf(v1, 0.f); }
      Cb[ci0] = v0;
      Cb[ci1] = v1;
    }
  } else {
    #pragma unroll
    for (int r = 0; r < 8; ++r) {
      const int gm = m0 + wm + r + lh * 8;
      if (gm >= M) continue;
      const float sc = rowScale ? rowScale[gm] : 1.f;
      const float sh = rowShift ? rowShift[gm] : 0.f;
      {
        const int gn = n0 + wn + lr;
        if (gn < N) {
          float v = acc0[r] * sc + sh;
          if (colBias) v += colBias[gn];
          const long ci = (long)gm * ldcM + gn;
          if (Rb) v += Rb[ci];
          if (relu) v = fmaxf(v, 0.f);
          Cb[ci] = v;
        }
      }
      {
        const int gn = n0 + wn + 16 + lr;
        if (gn < N) {
          float v = acc1[r] * sc + sh;
          if (colBias) v += colBias[gn];
          const long ci = (long)gm * ldcM + gn;
          if (Rb) v += Rb[ci];
          if (relu) v = fmaxf(v, 0.f);
          Cb[ci] = v;
        }
      }
    }
  }
}

// ------------------------- support kernels ---------------------------------

// im2col (1D): dst[b][(ci*kw+kk)][t] (bf16) from src (B,Ci,Tin) or channels-last
__global__ __launch_bounds__(256) void im2col_k(
    const float* __restrict__ src, u16* __restrict__ dst,
    int Ci, int Tin, int Tout, int kw, int stride, int pad,
    int chLast, int chN, long srcBatch, long dstBatch)
{
  const long total = (long)Ci * kw * Tout;
  const long idx = (long)blockIdx.x * 256 + threadIdx.x;
  if (idx >= total) return;
  const int b = blockIdx.y;
  const int t  = (int)(idx % Tout);
  const long ck = idx / Tout;
  const int kk = (int)(ck % kw);
  const int ci = (int)(ck / kw);
  const int pos = t * stride + kk - pad;
  float v = 0.f;
  if (pos >= 0 && pos < Tin)
    v = chLast ? src[(long)b * srcBatch + (long)pos * chN + ci]
               : src[(long)b * srcBatch + (long)ci * Tin + pos];
  dst[(long)b * dstBatch + idx] = f2bf(v);
}

__global__ __launch_bounds__(256) void cvt_bf16_k(const float* __restrict__ s,
                                                  u16* __restrict__ d, long n) {
  const long i = (long)blockIdx.x * 256 + threadIdx.x;
  if (i < n) d[i] = f2bf(s[i]);
}

// dst (C x R) bf16 = transpose of src (R x C) fp32
__global__ __launch_bounds__(256) void tcvt_k(const float* __restrict__ s,
                                              u16* __restrict__ d, int R, int Cc) {
  const long i = (long)blockIdx.x * 256 + threadIdx.x;
  if (i >= (long)R * Cc) return;
  const int c = (int)(i / R);
  const int r = (int)(i % R);
  d[i] = f2bf(s[(long)r * Cc + c]);
}

// (H,F,A) -> (F, H*A) bf16 : packed per-head projection matrix
__global__ __launch_bounds__(256) void reorder_hfa_k(const float* __restrict__ s,
                                                     u16* __restrict__ d,
                                                     int H, int F, int A) {
  const long i = (long)blockIdx.x * 256 + threadIdx.x;
  if (i >= (long)H * F * A) return;
  const int a = (int)(i % A);
  const long r = i / A;
  const int h = (int)(r % H);
  const int f = (int)(r / H);
  d[i] = f2bf(s[((long)h * F + f) * A + a]);
}

// BN fold: scale = g*rsqrt(var+eps); shift = (conv_b - mean)*scale + beta
__global__ __launch_bounds__(256) void bn_prep_k(const float* cb, const float* g,
                                                 const float* be, const float* mn,
                                                 const float* vr, float* scale,
                                                 float* shift, int C) {
  const int i = blockIdx.x * 256 + threadIdx.x;
  if (i >= C) return;
  const float sc = g[i] * rsqrtf(vr[i] + 1e-5f);
  scale[i] = sc;
  shift[i] = (cb[i] - mn[i]) * sc + be[i];
}

// (B,C,T) -> (B,T,C) fp32 + bf16
__global__ __launch_bounds__(256) void nct2ntc_k(const float* __restrict__ s,
                                                 float* __restrict__ dF,
                                                 u16* __restrict__ dB, int Cc, int T) {
  const long tot = (long)Cc * T;
  const long i = (long)blockIdx.x * 256 + threadIdx.x;
  if (i >= tot) return;
  const int b = blockIdx.y;
  const int c = (int)(i % Cc);
  const long t = i / Cc;
  const float v = s[(long)b * tot + (long)c * T + t];
  dF[(long)b * tot + i] = v;
  dB[(long)b * tot + i] = f2bf(v);
}

// logits = logits*invs + (|k-q|<=99 ? rlog[bh,q,k-q+99] : -1e8)
__global__ __launch_bounds__(256) void posbias_k(float* __restrict__ logits,
                                                 const float* __restrict__ rlog,
                                                 int T, int R, float invs) {
  const long idx = (long)blockIdx.x * 256 + threadIdx.x;
  const int k = (int)(idx % T);
  const long r1 = idx / T;
  const int q = (int)(r1 % T);
  const long bh = r1 / T;
  const int rel = k - q;
  float pos = -1.0e8f;
  if (rel >= -99 && rel <= 99)
    pos = rlog[bh * (long)T * R + (long)q * R + (rel + 99)];
  logits[idx] = logits[idx] * invs + pos;
}

// row softmax over T, writing bf16 probs (feeds next WMMA GEMM directly)
__global__ __launch_bounds__(256) void softmax_k(const float* __restrict__ logits,
                                                 u16* __restrict__ probs, int T) {
  __shared__ float red[256];
  const long row = blockIdx.x;
  const float* lp = logits + row * (long)T;
  u16* pp = probs + row * (long)T;
  const int tid = threadIdx.x;
  float mx = -3.4e38f;
  for (int i = tid; i < T; i += 256) mx = fmaxf(mx, lp[i]);
  red[tid] = mx; __syncthreads();
  for (int s = 128; s > 0; s >>= 1) { if (tid < s) red[tid] = fmaxf(red[tid], red[tid + s]); __syncthreads(); }
  mx = red[0]; __syncthreads();
  float sum = 0.f;
  for (int i = tid; i < T; i += 256) sum += __expf(lp[i] - mx);
  red[tid] = sum; __syncthreads();
  for (int s = 128; s > 0; s >>= 1) { if (tid < s) red[tid] += red[tid + s]; __syncthreads(); }
  const float inv = 1.0f / red[0];
  for (int i = tid; i < T; i += 256) pp[i] = f2bf(__expf(lp[i] - mx) * inv);
}

// layernorm over F features; outF fp32 (may be d_out), outB bf16 (optional)
__global__ __launch_bounds__(256) void layernorm_k(const float* __restrict__ x,
                                                   const float* __restrict__ g,
                                                   const float* __restrict__ b,
                                                   float* __restrict__ outF,
                                                   u16* __restrict__ outB, int F) {
  __shared__ float red[256];
  const long t = blockIdx.x;
  const float* xp = x + t * (long)F;
  const int tid = threadIdx.x;
  float s = 0.f;
  for (int i = tid; i < F; i += 256) s += xp[i];
  red[tid] = s; __syncthreads();
  for (int st = 128; st > 0; st >>= 1) { if (tid < st) red[tid] += red[tid + st]; __syncthreads(); }
  const float mu = red[0] / F; __syncthreads();
  float v = 0.f;
  for (int i = tid; i < F; i += 256) { const float d = xp[i] - mu; v += d * d; }
  red[tid] = v; __syncthreads();
  for (int st = 128; st > 0; st >>= 1) { if (tid < st) red[tid] += red[tid + st]; __syncthreads(); }
  const float rstd = rsqrtf(red[0] / F + 1e-5f);
  for (int i = tid; i < F; i += 256) {
    const float y = (xp[i] - mu) * rstd * g[i] + b[i];
    outF[t * (long)F + i] = y;
    if (outB) outB[t * (long)F + i] = f2bf(y);
  }
}

// ------------------------------- host --------------------------------------

static inline void launch_gemm(hipStream_t st,
    const u16* A, const u16* B, float* C, const float* Res,
    const float* rs, const float* rsh, const float* cb,
    int M, int N, int K,
    long ldaM, long ldbK, long ldbN, long ldcM,
    int Z, int H, long sAb, long sAh, long sBb, long sBh, long sCb, long sCh,
    int relu)
{
  dim3 grid((unsigned)((M + 63) / 64), (unsigned)((N + 63) / 64), (unsigned)Z);
  gemm_bf16_wmma<<<grid, 256, 0, st>>>(A, B, C, Res, rs, rsh, cb, M, N, K,
      ldaM, ldbK, ldbN, ldcM, H, sAb, sAh, sBb, sBh, sCb, sCh, relu);
}

static inline unsigned gblk(long n) { return (unsigned)((n + 255) / 256); }

extern "C" void kernel_launch(void* const* d_in, const int* in_sizes, int n_in,
                              void* d_out, int out_size, void* d_ws, size_t ws_size,
                              hipStream_t stream)
{
  (void)in_sizes; (void)out_size; (void)ws_size;
  if (n_in < 135) return;
  auto IN = [&](int i) { return (const float*)d_in[i]; };

  const int BB = 2, T = 1024, F = 768, NH = 8, DA = 32, DFF = 3072, NREL = 199;

  char* base = (char*)d_ws;
  size_t cur = 0;
  auto take = [&](size_t bytes) -> char* {
    char* p = base + cur; cur += (bytes + 255) & ~(size_t)255; return p;
  };

  // Shared slab: conv arena (94.4MB) and attention arena (113.7MB) are
  // time-disjoint (attention buffers first written after conv outputs consumed).
  const size_t SLAB = 113704960;
  char* slab = take(SLAB);
  u16*   colA   = (u16*)(slab);                 // im2col (max 2*2304*4096 bf16)
  u16*   colB   = (u16*)(slab + 37748736);      // 1x1-res im2col
  float* bufA   = (float*)(slab + 44040192);    // conv h
  float* bufB   = (float*)(slab + 69206016);    // block in/out
  float* logits = (float*)(slab);               // (2,8,1024,1024) f32
  u16*   probsb = (u16*)(slab + 67108864);      // (2,8,1024,1024) bf16
  float* rlog   = (float*)(slab + 100663296);   // (2,8,1024,199) f32

  float* xbuf   = (float*)take((size_t)BB * T * F * 4);
  float* xbuf2  = (float*)take((size_t)BB * T * F * 4);
  u16*   xbf    = (u16*)take((size_t)BB * T * F * 2);
  float* Qf = (float*)take((size_t)BB * T * NH * DA * 4);
  float* Kf = (float*)take((size_t)BB * T * NH * DA * 4);
  float* Vf = (float*)take((size_t)BB * T * NH * DA * 4);
  u16* Qb = (u16*)take((size_t)BB * T * NH * DA * 2);
  u16* Kb = (u16*)take((size_t)BB * T * NH * DA * 2);
  u16* Vb = (u16*)take((size_t)BB * T * NH * DA * 2);
  float* of = (float*)take((size_t)BB * T * NH * DA * 4);
  u16*   ob = (u16*)take((size_t)BB * T * NH * DA * 2);
  float* ffh  = (float*)take((size_t)BB * T * DFF * 4);
  u16*   ffhb = (u16*)take((size_t)BB * T * DFF * 2);
  u16*   wslot   = (u16*)take((size_t)F * DFF * 2);
  u16*   relslot = (u16*)take((size_t)NH * NREL * DA * 2);
  float* scaleB = (float*)take(4096);
  float* shiftB = (float*)take(4096);

  // ---------------- conv resblocks (conv-as-GEMM via im2col) ---------------
  struct Cfg { int ci, co, Tin, Tout; };
  const Cfg cfgs[3] = { {8, F, 8192, 4096}, {F, F, 4096, 2048}, {F, F, 2048, 1024} };
  const float* srcP = IN(0);
  int chLast = 1, chN = 8;
  long srcBatch = 8192L * 8;

  for (int bk = 0; bk < 3; ++bk) {
    const Cfg c = cfgs[bk];
    const int wb = 1 + 18 * bk;
    const int K1 = c.ci * 3;
    // conv1 (stride 2, pad 1): im2col -> GEMM with fused BN1+ReLU
    im2col_k<<<dim3(gblk((long)K1 * c.Tout), 2, 1), 256, 0, stream>>>(
        srcP, colA, c.ci, c.Tin, c.Tout, 3, 2, 1, chLast, chN, srcBatch, (long)K1 * c.Tout);
    cvt_bf16_k<<<gblk((long)c.co * K1), 256, 0, stream>>>(IN(wb + 0), wslot, (long)c.co * K1);
    bn_prep_k<<<gblk(c.co), 256, 0, stream>>>(IN(wb + 1), IN(wb + 2), IN(wb + 3),
                                              IN(wb + 4), IN(wb + 5), scaleB, shiftB, c.co);
    launch_gemm(stream, wslot, colA, bufA, nullptr, scaleB, shiftB, nullptr,
                c.co, c.Tout, K1, K1, c.Tout, 1, c.Tout,
                2, 1, 0, 0, (long)K1 * c.Tout, 0, (long)c.co * c.Tout, 0, 1);
    // conv2 im2col (stride 1, pad 1) from bufA
    const int K2 = c.co * 3;
    im2col_k<<<dim3(gblk((long)K2 * c.Tout), 2, 1), 256, 0, stream>>>(
        bufA, colA, c.co, c.Tout, c.Tout, 3, 1, 1, 0, 0, (long)c.co * c.Tout, (long)K2 * c.Tout);
    // residual 1x1 (stride 2) im2col from block input
    im2col_k<<<dim3(gblk((long)c.ci * c.Tout), 2, 1), 256, 0, stream>>>(
        srcP, colB, c.ci, c.Tin, c.Tout, 1, 2, 0, chLast, chN, srcBatch, (long)c.ci * c.Tout);
    // residual branch: res_bn(conv1x1(x)) -> bufB (overwrites consumed input)
    cvt_bf16_k<<<gblk((long)c.co * c.ci), 256, 0, stream>>>(IN(wb + 12), wslot, (long)c.co * c.ci);
    bn_prep_k<<<gblk(c.co), 256, 0, stream>>>(IN(wb + 13), IN(wb + 14), IN(wb + 15),
                                              IN(wb + 16), IN(wb + 17), scaleB, shiftB, c.co);
    launch_gemm(stream, wslot, colB, bufB, nullptr, scaleB, shiftB, nullptr,
                c.co, c.Tout, c.ci, c.ci, c.Tout, 1, c.Tout,
                2, 1, 0, 0, (long)c.ci * c.Tout, 0, (long)c.co * c.Tout, 0, 0);
    // conv2: BN2 + residual(bufB) + ReLU fused
    cvt_bf16_k<<<gblk((long)c.co * K2), 256, 0, stream>>>(IN(wb + 6), wslot, (long)c.co * K2);
    bn_prep_k<<<gblk(c.co), 256, 0, stream>>>(IN(wb + 7), IN(wb + 8), IN(wb + 9),
                                              IN(wb + 10), IN(wb + 11), scaleB, shiftB, c.co);
    launch_gemm(stream, wslot, colA, bufB, bufB, scaleB, shiftB, nullptr,
                c.co, c.Tout, K2, K2, c.Tout, 1, c.Tout,
                2, 1, 0, 0, (long)K2 * c.Tout, 0, (long)c.co * c.Tout, 0, 1);
    srcP = bufB; chLast = 0; chN = 0; srcBatch = (long)c.co * c.Tout;
  }

  // (B,768,1024) -> tokens (B,1024,768) fp32+bf16
  nct2ntc_k<<<dim3(gblk((long)F * T), 2, 1), 256, 0, stream>>>(bufB, xbuf, xbf, F, T);

  // input projection: x @ w_in^T + b_in
  tcvt_k<<<gblk((long)F * F), 256, 0, stream>>>(IN(55), wslot, F, F);
  launch_gemm(stream, xbf, wslot, xbuf, nullptr, nullptr, nullptr, IN(56),
              BB * T, F, F, F, F, 1, F, 1, 1, 0, 0, 0, 0, 0, 0, 0);
  cvt_bf16_k<<<gblk((long)BB * T * F), 256, 0, stream>>>(xbuf, xbf, (long)BB * T * F);

  // ------------------------- transformer layers ----------------------------
  const long nQ = (long)BB * T * NH * DA;
  for (int l = 0; l < 6; ++l) {
    const int li = 57 + 13 * l;
    const float* Wq = IN(li), *Wk = IN(li + 1), *Wv = IN(li + 2), *Wo = IN(li + 3), *REL = IN(li + 4);
    const float* L1G = IN(li + 5), *L1B = IN(li + 6);
    const float* W1 = IN(li + 7), *B1 = IN(li + 8), *W2 = IN(li + 9), *B2 = IN(li + 10);
    const float* L2G = IN(li + 11), *L2B = IN(li + 12);

    // Q/K/V projections (2048x768)x(768x256)
    reorder_hfa_k<<<gblk((long)NH * F * DA), 256, 0, stream>>>(Wq, wslot, NH, F, DA);
    launch_gemm(stream, xbf, wslot, Qf, nullptr, nullptr, nullptr, nullptr,
                BB * T, NH * DA, F, F, NH * DA, 1, NH * DA, 1, 1, 0,0,0,0,0,0, 0);
    cvt_bf16_k<<<gblk(nQ), 256, 0, stream>>>(Qf, Qb, nQ);
    reorder_hfa_k<<<gblk((long)NH * F * DA), 256, 0, stream>>>(Wk, wslot, NH, F, DA);
    launch_gemm(stream, xbf, wslot, Kf, nullptr, nullptr, nullptr, nullptr,
                BB * T, NH * DA, F, F, NH * DA, 1, NH * DA, 1, 1, 0,0,0,0,0,0, 0);
    cvt_bf16_k<<<gblk(nQ), 256, 0, stream>>>(Kf, Kb, nQ);
    reorder_hfa_k<<<gblk((long)NH * F * DA), 256, 0, stream>>>(Wv, wslot, NH, F, DA);
    launch_gemm(stream, xbf, wslot, Vf, nullptr, nullptr, nullptr, nullptr,
                BB * T, NH * DA, F, F, NH * DA, 1, NH * DA, 1, 1, 0,0,0,0,0,0, 0);
    cvt_bf16_k<<<gblk(nQ), 256, 0, stream>>>(Vf, Vb, nQ);

    // logits[b,h] = Q_bh (1024x32) x K_bh^T (32x1024)
    launch_gemm(stream, Qb, Kb, logits, nullptr, nullptr, nullptr, nullptr,
                T, T, DA, (long)NH * DA, 1, (long)NH * DA, T,
                BB * NH, NH, (long)T * NH * DA, DA, (long)T * NH * DA, DA,
                (long)NH * T * T, (long)T * T, 0);

    // rlog[b,h] = Q_bh (1024x32) x rel_h^T (32x199)
    cvt_bf16_k<<<gblk((long)NH * NREL * DA), 256, 0, stream>>>(REL, relslot, (long)NH * NREL * DA);
    launch_gemm(stream, Qb, relslot, rlog, nullptr, nullptr, nullptr, nullptr,
                T, NREL, DA, (long)NH * DA, 1, DA, NREL,
                BB * NH, NH, (long)T * NH * DA, DA, 0, (long)NREL * DA,
                (long)NH * T * NREL, (long)T * NREL, 0);

    // scale + gather relative bias + band mask; then softmax -> bf16 probs
    posbias_k<<<gblk((long)BB * NH * T * T), 256, 0, stream>>>(logits, rlog, T, NREL,
                                                               0.17677669529663687f);
    softmax_k<<<BB * NH * T, 256, 0, stream>>>(logits, probsb, T);

    // o[b,h] = probs (1024x1024) x V_bh (1024x32), stored (b,t,h,a)
    launch_gemm(stream, probsb, Vb, of, nullptr, nullptr, nullptr, nullptr,
                T, DA, T, T, (long)NH * DA, 1, (long)NH * DA,
                BB * NH, NH, (long)NH * T * T, (long)T * T,
                (long)T * NH * DA, DA, (long)T * NH * DA, DA, 0);
    cvt_bf16_k<<<gblk(nQ), 256, 0, stream>>>(of, ob, nQ);

    // attn out: (2048x256)x(256x768) + residual x; then LN1
    cvt_bf16_k<<<gblk((long)NH * DA * F), 256, 0, stream>>>(Wo, wslot, (long)NH * DA * F);
    launch_gemm(stream, ob, wslot, xbuf2, xbuf, nullptr, nullptr, nullptr,
                BB * T, F, NH * DA, NH * DA, F, 1, F, 1, 1, 0,0,0,0,0,0, 0);
    layernorm_k<<<BB * T, 256, 0, stream>>>(xbuf2, L1G, L1B, xbuf, xbf, F);

    // FFN: relu(x@w1+b1) @ w2 + b2 + residual; then LN2
    cvt_bf16_k<<<gblk((long)F * DFF), 256, 0, stream>>>(W1, wslot, (long)F * DFF);
    launch_gemm(stream, xbf, wslot, ffh, nullptr, nullptr, nullptr, B1,
                BB * T, DFF, F, F, DFF, 1, DFF, 1, 1, 0,0,0,0,0,0, 1);
    cvt_bf16_k<<<gblk((long)BB * T * DFF), 256, 0, stream>>>(ffh, ffhb, (long)BB * T * DFF);
    cvt_bf16_k<<<gblk((long)DFF * F), 256, 0, stream>>>(W2, wslot, (long)DFF * F);
    launch_gemm(stream, ffhb, wslot, xbuf2, xbuf, nullptr, nullptr, B2,
                BB * T, F, DFF, DFF, F, 1, F, 1, 1, 0,0,0,0,0,0, 0);

    float* lnout = (l == 5) ? (float*)d_out : xbuf;
    u16* lnoutb = (l == 5) ? (u16*)nullptr : xbf;
    layernorm_k<<<BB * T, 256, 0, stream>>>(xbuf2, L2G, L2B, lnout, lnoutb, F);
  }
}
